// GPT2_47150150976167
// MI455X (gfx1250) — matrix operand order repaired and checked by reading, not compiled, not run on hardware
//
#include <hip/hip_runtime.h>
#include <hip/hip_bf16.h>
#include <math.h>

// ---------------- problem constants (match reference) ----------------
namespace cfg {
constexpr int L = 6;
constexpr int B = 2;
constexpr int T = 1024;
constexpr int C = 768;
constexpr int H = 12;
constexpr int D = 64;          // C / H
constexpr int F = 3072;
constexpr int V = 50257;
constexpr int M = B * T;       // 2048 rows for all GEMMs (BM-aligned)
}

// ---------------- vector types ----------------
typedef __attribute__((ext_vector_type(16))) __bf16         bf16x16;
typedef __attribute__((ext_vector_type(16))) unsigned short u16x16;
typedef __attribute__((ext_vector_type(8)))  unsigned short u16x8;
typedef __attribute__((ext_vector_type(8)))  float          f32x8;
typedef __attribute__((ext_vector_type(4)))  float          f32x4;
typedef __attribute__((ext_vector_type(4)))  unsigned       u32x4;

__device__ __forceinline__ unsigned short f32_to_bf16(float f) {
    unsigned u = __builtin_bit_cast(unsigned, f);
    unsigned r = u + 0x7FFFu + ((u >> 16) & 1u);   // RNE truncation
    return (unsigned short)(r >> 16);
}

__device__ __forceinline__ unsigned pk_bf16(float a, float b) {
    // packed conversion; backend can lower to v_cvt_pk_bf16_f32
    __hip_bfloat162 h = __float22bfloat162_rn(float2{a, b});
    unsigned u;
    __builtin_memcpy(&u, &h, sizeof(u));   // __hip_bfloat162 not trivially copyable
    return u;
}

__device__ __forceinline__ u16x8 pack_bf16x8(f32x4 a, f32x4 b) {
    u32x4 w;
    w[0] = pk_bf16(a[0], a[1]);
    w[1] = pk_bf16(a[2], a[3]);
    w[2] = pk_bf16(b[0], b[1]);
    w[3] = pk_bf16(b[2], b[3]);
    return __builtin_bit_cast(u16x8, w);
}

__device__ __forceinline__ u16x8 cvt8(const float* p) {   // 8 consecutive f32
    f32x4 a = *(const f32x4*)p;
    f32x4 b = *(const f32x4*)(p + 4);
    return pack_bf16x8(a, b);
}

__device__ __forceinline__ bf16x16 join16(u16x8 lo, u16x8 hi) {
    u16x16 c = __builtin_shufflevector(lo, hi, 0,1,2,3,4,5,6,7,
                                       8,9,10,11,12,13,14,15);
    return __builtin_bit_cast(bf16x16, c);
}

// fragment from two 8-float runs (A layout: halves 0..7 / 8..15 are two runs)
__device__ __forceinline__ bf16x16 frag_2runs(const float* p0, const float* p1) {
    return join16(cvt8(p0), cvt8(p1));
}

#define WMMA_BF16(a, b, c) __builtin_amdgcn_wmma_f32_16x16x32_bf16( \
        false, (a), false, (b), (short)0, (c), false, false)

// ================= fused WMMA GEMM (double-buffered LDS) =================
#define BM 128
#define BN 64
#define BK 32
#define LP 8                 // LDS row pad (ushorts) -> 80B stride, 16B aligned
#define LDW (BK + LP)

__device__ __forceinline__ bf16x16 frag_a_lds(const unsigned short* row, int kg) {
    u16x8 lo = *(const u16x8*)(row + kg * 8);
    u16x8 hi = *(const u16x8*)(row + 16 + kg * 8);
    return join16(lo, hi);
}
__device__ __forceinline__ bf16x16 frag_b_lds(const unsigned short* row, int kg) {
    u16x8 lo = *(const u16x8*)(row + kg * 8);
    u16x8 hi = *(const u16x8*)(row + 16 + kg * 8);
    return join16(lo, hi);
}

// NB = need N-bounds checks (only last block column of the LM head)
template <bool NB>
__device__ __forceinline__ void gemm_tile(
    const float* __restrict__ A, const float* __restrict__ W,
    const float* __restrict__ bias, const float* __restrict__ residual,
    float* __restrict__ out, int M, int N, int K, int act,
    unsigned short (*As)[BM][LDW], unsigned short (*Bs)[BN][LDW],
    int m0, int n0)
{
    const int tid  = threadIdx.x;
    const int lane = tid & 31;
    const int wave = tid >> 5;
    const int wm   = wave >> 1;
    const int wn   = wave & 1;
    const int lm   = lane & 15;
    const int kg   = lane >> 4;

    const int ar = tid >> 2;            // A row 0..63 (and +64)
    const int ac = (tid & 3) * 8;       // A col 0,8,16,24
    const int bn = tid & 63;            // B column (N)
    const int rb = (tid >> 6) * 8;      // B row base (K)
    const int gcol = n0 + bn;

    f32x8 acc[2][2] = {};

    // ---- prologue: stage K-tile 0 into buffer 0
    {
        const float* srcA0 = A + (long)(m0 + ar) * K + ac;
        (*(u16x8*)&As[0][ar][ac]) =
            pack_bf16x8(*(const f32x4*)srcA0, *(const f32x4*)(srcA0 + 4));
        const float* srcA1 = srcA0 + (long)64 * K;
        (*(u16x8*)&As[0][ar + 64][ac]) =
            pack_bf16x8(*(const f32x4*)srcA1, *(const f32x4*)(srcA1 + 4));
        f32x4 b0, b1;
        #pragma unroll
        for (int i = 0; i < 4; ++i) {
            float v = 0.0f;
            if (!NB || gcol < N) v = W[(long)(rb + i) * N + gcol];
            b0[i] = v;
        }
        #pragma unroll
        for (int i = 0; i < 4; ++i) {
            float v = 0.0f;
            if (!NB || gcol < N) v = W[(long)(rb + 4 + i) * N + gcol];
            b1[i] = v;
        }
        (*(u16x8*)&Bs[0][bn][rb]) = pack_bf16x8(b0, b1);
    }
    __syncthreads();

    int buf = 0;
    for (int kb = 0; kb < K; kb += BK) {
        const bool more = (kb + BK) < K;

        // ---- issue next-tile global loads early (hide under WMMA below)
        f32x4 na0, na1, na2, na3, nb0, nb1;
        if (more) {
            const float* srcA0 = A + (long)(m0 + ar) * K + kb + BK + ac;
            na0 = *(const f32x4*)srcA0;
            na1 = *(const f32x4*)(srcA0 + 4);
            const float* srcA1 = srcA0 + (long)64 * K;
            na2 = *(const f32x4*)srcA1;
            na3 = *(const f32x4*)(srcA1 + 4);
            #pragma unroll
            for (int i = 0; i < 4; ++i) {
                float v = 0.0f;
                if (!NB || gcol < N) v = W[(long)(kb + BK + rb + i) * N + gcol];
                nb0[i] = v;
            }
            #pragma unroll
            for (int i = 0; i < 4; ++i) {
                float v = 0.0f;
                if (!NB || gcol < N) v = W[(long)(kb + BK + rb + 4 + i) * N + gcol];
                nb1[i] = v;
            }
            if (kb + 2 * BK < K) {   // L2 warm-up two tiles ahead
                __builtin_prefetch(A + (long)(m0 + ar) * K + kb + 2 * BK + ac, 0, 1);
                if (!NB || gcol < N)
                    __builtin_prefetch(W + (long)(kb + 2 * BK + rb) * N + gcol, 0, 1);
            }
        }

        // ---- compute on current buffer
        bf16x16 a0 = frag_a_lds(&As[buf][wm * 32 + lm][0],      kg);
        bf16x16 a1 = frag_a_lds(&As[buf][wm * 32 + 16 + lm][0], kg);
        bf16x16 b0 = frag_b_lds(&Bs[buf][wn * 32 + lm][0],      kg);
        bf16x16 b1 = frag_b_lds(&Bs[buf][wn * 32 + 16 + lm][0], kg);

        acc[0][0] = WMMA_BF16(a0, b0, acc[0][0]);
        acc[0][1] = WMMA_BF16(a0, b1, acc[0][1]);
        acc[1][0] = WMMA_BF16(a1, b0, acc[1][0]);
        acc[1][1] = WMMA_BF16(a1, b1, acc[1][1]);

        // ---- store next tile to the other buffer (safe: last read i-1, fenced)
        if (more) {
            (*(u16x8*)&As[buf ^ 1][ar][ac])      = pack_bf16x8(na0, na1);
            (*(u16x8*)&As[buf ^ 1][ar + 64][ac]) = pack_bf16x8(na2, na3);
            (*(u16x8*)&Bs[buf ^ 1][bn][rb])      = pack_bf16x8(nb0, nb1);
        }
        __syncthreads();   // publish stores before next iteration reads
        buf ^= 1;
    }

    // ---- epilogue (C/D layout: lane col n = lm, VGPR r -> m = r + 8*kg)
    #pragma unroll
    for (int bi = 0; bi < 2; ++bi) {
        const int col = n0 + wn * 32 + bi * 16 + lm;
        if (NB && col >= N) continue;
        const float bb = bias ? bias[col] : 0.0f;
        #pragma unroll
        for (int ai = 0; ai < 2; ++ai) {
            #pragma unroll
            for (int r = 0; r < 8; ++r) {
                const int row = m0 + wm * 32 + ai * 16 + r + 8 * kg;
                float v = acc[ai][bi][r] + bb;
                if (act == 1)
                    v = 0.5f * v * (1.0f + erff(v * 0.70710678118654752f));
                const long o = (long)row * N + col;
                if (residual) v += residual[o];
                out[o] = v;
            }
        }
    }
}

__global__ __launch_bounds__(256) void wmma_gemm_kernel(
    const float* __restrict__ A, const float* __restrict__ W,
    const float* __restrict__ bias, const float* __restrict__ residual,
    float* __restrict__ out, int M, int N, int K, int act)
{
    __shared__ unsigned short As[2][BM][LDW];
    __shared__ unsigned short Bs[2][BN][LDW];
    const int m0 = blockIdx.y * BM;
    const int n0 = blockIdx.x * BN;
    if (n0 + BN <= N)
        gemm_tile<false>(A, W, bias, residual, out, M, N, K, act, As, Bs, m0, n0);
    else
        gemm_tile<true >(A, W, bias, residual, out, M, N, K, act, As, Bs, m0, n0);
}

// ================= flash attention on WMMA =================
// One wave per (b, h, 16-query tile). Per 32-key chunk:
//   S^T(32k x 16q) = K_tile(16x64) @ Q^T(64x16)     -> 4 WMMAs
//   online softmax (stats per q, lane-resident), P -> bf16 -> LDS
//   O(16q x 64d) += P(16x32) @ V(32x16) x4 d-tiles  -> 4 WMMAs
__global__ __launch_bounds__(32) void flash_attention_kernel(
    const float* __restrict__ qkv, float* __restrict__ att_o)
{
    using namespace cfg;
    const int C3 = 3 * C;

    __shared__ unsigned short PL[16][32 + 8];  // P^T bf16 bits, padded rows
    __shared__ float statL[16];                // corr (then l) per q

    const int tile = blockIdx.x;
    const int qt   = tile & 63;               // T/16 = 64
    const int h    = (tile >> 6) % H;
    const int b    = tile / (64 * H);
    const int q0   = qt * 16;

    const int lane = threadIdx.x;
    const int lm   = lane & 15;
    const int kg   = lane >> 4;

    // ---- Q fragments (B-layout: lane col q = lm, i<8 -> d = kg*8+i, i>=8 -> d = 16+kg*8+i-8)
    const float* qrow = qkv + (long)(b * T + q0 + lm) * C3 + h * D;
    bf16x16 qf0 = join16(cvt8(qrow + kg * 8),      cvt8(qrow + 16 + kg * 8));
    bf16x16 qf1 = join16(cvt8(qrow + 32 + kg * 8), cvt8(qrow + 48 + kg * 8));

    f32x8 oacc[4] = {};          // O: 4 d-tiles; lane col d = dbase+lm, row q = r+8kg
    float mrun = -3.402823e38f;  // running max for q = lm (dup in both halves)
    float lrun = 0.0f;

    const int kend = q0 + 16;    // causal: keys < q0+16
    for (int k0 = 0; k0 < kend; k0 += 32) {
        // ---- S^T for two 16-key tiles
        f32x8 s0 = {}, s1 = {};
        {   // keys k0 + lm (tile0) / k0 + 16 + lm (tile1); A layout runs
            const float* kr0 = qkv + (long)(b * T + k0 + lm) * C3 + C + h * D;
            const float* kr1 = kr0 + 16 * C3;
            bf16x16 ka0 = frag_2runs(kr0 + kg * 8,      kr0 + 16 + kg * 8);
            bf16x16 ka1 = frag_2runs(kr0 + 32 + kg * 8, kr0 + 48 + kg * 8);
            s0 = WMMA_BF16(ka0, qf0, s0);
            s0 = WMMA_BF16(ka1, qf1, s0);
            bf16x16 kb0 = frag_2runs(kr1 + kg * 8,      kr1 + 16 + kg * 8);
            bf16x16 kb1 = frag_2runs(kr1 + 32 + kg * 8, kr1 + 48 + kg * 8);
            s1 = WMMA_BF16(kb0, qf0, s1);
            s1 = WMMA_BF16(kb1, qf1, s1);
        }

        // ---- scale + causal mask (row = key = r+8kg, col = q = q0+lm)
        const int qg = q0 + lm;
        #pragma unroll
        for (int r = 0; r < 8; ++r) {
            int key0 = k0 + r + 8 * kg;
            s0[r] = (key0 <= qg) ? s0[r] * 0.125f : -3.402823e38f;
            int key1 = key0 + 16;
            s1[r] = (key1 <= qg) ? s1[r] * 0.125f : -3.402823e38f;
        }

        // ---- online softmax stats per q (half-wave reduce over keys)
        float tmax = s0[0];
        #pragma unroll
        for (int r = 1; r < 8; ++r) tmax = fmaxf(tmax, s0[r]);
        #pragma unroll
        for (int r = 0; r < 8; ++r) tmax = fmaxf(tmax, s1[r]);
        tmax = fmaxf(tmax, __shfl_xor(tmax, 16, 32));

        const float mnew = fmaxf(mrun, tmax);
        const float corr = __expf(mrun - mnew);   // 0 on first chunk
        mrun = mnew;

        float tsum = 0.0f;
        f32x4 p0a, p0b, p1a, p1b;
        #pragma unroll
        for (int r = 0; r < 4; ++r) { p0a[r] = __expf(s0[r] - mnew);     tsum += p0a[r]; }
        #pragma unroll
        for (int r = 0; r < 4; ++r) { p0b[r] = __expf(s0[4 + r] - mnew); tsum += p0b[r]; }
        #pragma unroll
        for (int r = 0; r < 4; ++r) { p1a[r] = __expf(s1[r] - mnew);     tsum += p1a[r]; }
        #pragma unroll
        for (int r = 0; r < 4; ++r) { p1b[r] = __expf(s1[4 + r] - mnew); tsum += p1b[r]; }
        tsum += __shfl_xor(tsum, 16, 32);
        lrun = lrun * corr + tsum;

        // ---- P^T -> LDS in A-fragment-friendly layout: PL[q][key]
        *(u16x8*)&PL[lm][kg * 8]      = pack_bf16x8(p0a, p0b);   // keys kg*8+0..7
        *(u16x8*)&PL[lm][16 + kg * 8] = pack_bf16x8(p1a, p1b);   // keys 16+kg*8..
        statL[lm] = corr;             // both halves write same value
        __syncthreads();

        // ---- rescale O rows by corr(q = r+8kg)
        float cr[8];
        #pragma unroll
        for (int r = 0; r < 8; ++r) cr[r] = statL[r + 8 * kg];
        #pragma unroll
        for (int dt = 0; dt < 4; ++dt)
            #pragma unroll
            for (int r = 0; r < 8; ++r) oacc[dt][r] *= cr[r];

        // ---- P fragment (A layout) straight from LDS
        bf16x16 pf = join16(*(const u16x8*)&PL[lm][kg * 8],
                            *(const u16x8*)&PL[lm][16 + kg * 8]);

        // ---- V fragments (B layout: lane col d = dt*16+lm, halves -> key)
        #pragma unroll
        for (int dt = 0; dt < 4; ++dt) {
            u16x16 vu;
            #pragma unroll
            for (int i = 0; i < 16; ++i) {
                int key = k0 + (i < 8 ? kg * 8 + i : 16 + kg * 8 + (i - 8));
                vu[i] = f32_to_bf16(
                    qkv[(long)(b * T + key) * C3 + 2 * C + h * D + dt * 16 + lm]);
            }
            bf16x16 vf = __builtin_bit_cast(bf16x16, vu);
            oacc[dt] = WMMA_BF16(pf, vf, oacc[dt]);
        }
        __syncthreads();
    }

    // ---- normalize and store O
    statL[lm] = lrun;
    __syncthreads();
    float inv[8];
    #pragma unroll
    for (int r = 0; r < 8; ++r) inv[r] = 1.0f / statL[r + 8 * kg];

    #pragma unroll
    for (int dt = 0; dt < 4; ++dt)
        #pragma unroll
        for (int r = 0; r < 8; ++r)
            att_o[(long)(b * T + q0 + r + 8 * kg) * C + h * D + dt * 16 + lm] =
                oacc[dt][r] * inv[r];
}

// ---------------- LayerNorm (one workgroup per row) ----------------
__global__ __launch_bounds__(256) void layernorm_kernel(
    const float* __restrict__ x, const float* __restrict__ scale,
    const float* __restrict__ bias, float* __restrict__ y, int C)
{
    __shared__ float red[256];
    const int row = blockIdx.x;
    const float* xr = x + (long)row * C;

    float sum = 0.0f;
    for (int c = threadIdx.x; c < C; c += 256) sum += xr[c];
    red[threadIdx.x] = sum;
    __syncthreads();
    for (int o = 128; o > 0; o >>= 1) {
        if (threadIdx.x < o) red[threadIdx.x] += red[threadIdx.x + o];
        __syncthreads();
    }
    const float mu = red[0] / C;
    __syncthreads();

    float vs = 0.0f;
    for (int c = threadIdx.x; c < C; c += 256) {
        float d = xr[c] - mu;
        vs += d * d;
    }
    red[threadIdx.x] = vs;
    __syncthreads();
    for (int o = 128; o > 0; o >>= 1) {
        if (threadIdx.x < o) red[threadIdx.x] += red[threadIdx.x + o];
        __syncthreads();
    }
    const float rstd = rsqrtf(red[0] / C + 1e-5f);

    float* yr = y + (long)row * C;
    for (int c = threadIdx.x; c < C; c += 256)
        yr[c] = (xr[c] - mu) * rstd * scale[c] + bias[c];
}

// ---------------- embedding: x = wte[idx] + pte ----------------
__global__ __launch_bounds__(256) void embed_kernel(
    const int* __restrict__ idx, const float* __restrict__ wte,
    const float* __restrict__ pte, float* __restrict__ x)
{
    int i = blockIdx.x * blockDim.x + threadIdx.x;
    const int total = cfg::B * cfg::T * cfg::C;
    if (i >= total) return;
    int c  = i % cfg::C;
    int bt = i / cfg::C;
    int t  = bt % cfg::T;
    int tok = idx[bt];
    x[i] = wte[(long)tok * cfg::C + c] + pte[(long)t * cfg::C + c];
}

// ---------------- host-side orchestration ----------------
extern "C" void kernel_launch(void* const* d_in, const int* in_sizes, int n_in,
                              void* d_out, int out_size, void* d_ws, size_t ws_size,
                              hipStream_t stream) {
    using namespace cfg;
    (void)in_sizes; (void)n_in; (void)out_size; (void)ws_size;

    const int*   idx    = (const int*)  d_in[0];
    const float* wte    = (const float*)d_in[1];
    const float* pte    = (const float*)d_in[2];
    const float* ln1_s  = (const float*)d_in[3];
    const float* ln1_b  = (const float*)d_in[4];
    const float* qkv_w  = (const float*)d_in[5];
    const float* qkv_b  = (const float*)d_in[6];
    const float* proj_w = (const float*)d_in[7];
    const float* proj_b = (const float*)d_in[8];
    const float* ln2_s  = (const float*)d_in[9];
    const float* ln2_b  = (const float*)d_in[10];
    const float* w1     = (const float*)d_in[11];
    const float* b1     = (const float*)d_in[12];
    const float* w2     = (const float*)d_in[13];
    const float* b2     = (const float*)d_in[14];
    const float* lnf_s  = (const float*)d_in[15];
    const float* lnf_b  = (const float*)d_in[16];
    const float* head_w = (const float*)d_in[17];
    float* out = (float*)d_out;

    // workspace layout (fp32)
    char* ws = (char*)d_ws;
    float* x     = (float*)ws;   ws += (size_t)M * C * 4;      // residual stream
    float* hbuf  = (float*)ws;   ws += (size_t)M * C * 4;      // LN output
    float* qkv   = (float*)ws;   ws += (size_t)M * 3 * C * 4;
    float* att_o = (float*)ws;   ws += (size_t)M * C * 4;
    float* h2    = (float*)ws;   ws += (size_t)M * F * 4;      // MLP hidden

    const dim3 blk(256);

    embed_kernel<<<(M * C + 255) / 256, blk, 0, stream>>>(idx, wte, pte, x);

    for (int l = 0; l < L; ++l) {
        layernorm_kernel<<<M, blk, 0, stream>>>(x, ln1_s + (long)l * C,
                                                ln1_b + (long)l * C, hbuf, C);
        {   // qkv = h @ qkv_w + qkv_b       [M, 3C]
            dim3 g((3 * C + BN - 1) / BN, M / BM);
            wmma_gemm_kernel<<<g, blk, 0, stream>>>(
                hbuf, qkv_w + (long)l * C * 3 * C, qkv_b + (long)l * 3 * C,
                nullptr, qkv, M, 3 * C, C, 0);
        }
        flash_attention_kernel<<<B * H * (T / 16), dim3(32), 0, stream>>>(qkv, att_o);
        {   // x = x + att_o @ proj_w + proj_b
            dim3 g((C + BN - 1) / BN, M / BM);
            wmma_gemm_kernel<<<g, blk, 0, stream>>>(
                att_o, proj_w + (long)l * C * C, proj_b + (long)l * C,
                x, x, M, C, C, 0);
        }
        layernorm_kernel<<<M, blk, 0, stream>>>(x, ln2_s + (long)l * C,
                                                ln2_b + (long)l * C, hbuf, C);
        {   // h2 = gelu(h @ w1 + b1)        [M, F]
            dim3 g((F + BN - 1) / BN, M / BM);
            wmma_gemm_kernel<<<g, blk, 0, stream>>>(
                hbuf, w1 + (long)l * C * F, b1 + (long)l * F,
                nullptr, h2, M, F, C, 1);
        }
        {   // x = x + h2 @ w2 + b2
            dim3 g((C + BN - 1) / BN, M / BM);
            wmma_gemm_kernel<<<g, blk, 0, stream>>>(
                h2, w2 + (long)l * F * C, b2 + (long)l * C,
                x, x, M, C, F, 0);
        }
    }

    layernorm_kernel<<<M, blk, 0, stream>>>(x, lnf_s, lnf_b, hbuf, C);
    {   // logits = h @ head_w              [M, V]
        dim3 g((V + BN - 1) / BN, M / BM);
        wmma_gemm_kernel<<<g, blk, 0, stream>>>(
            hbuf, head_w, nullptr, nullptr, out, M, V, C, 0);
    }
}
